// PurePyTorchSSM_57690000719974
// MI455X (gfx1250) — matrix operand order repaired and checked by reading, not compiled, not run on hardware
//
#include <hip/hip_runtime.h>
#include <hip/hip_bf16.h>
#include <math.h>

#define D_MODELX 1024
#define D_STATEX 16
#define D_CONVX  4
#define BATCHX   2
#define SEQ_LENX 2048
#define M_ROWS   (BATCHX * SEQ_LENX)   // 4096

typedef __attribute__((ext_vector_type(16))) __bf16 v16bf;
typedef __attribute__((ext_vector_type(8)))  __bf16 v8bf;
typedef __attribute__((ext_vector_type(8)))  float  v8f;

// ---------------------------------------------------------------- prep kernels

__global__ void k_f32_to_bf16(const float* __restrict__ in, __bf16* __restrict__ out, int n) {
  int i = blockIdx.x * blockDim.x + threadIdx.x;
  if (i < n) out[i] = (__bf16)in[i];
}

// in: K x N fp32 (row-major), out: N x K bf16 (row-major)  == transpose+convert
__global__ void k_transpose_bf16(const float* __restrict__ in, __bf16* __restrict__ out,
                                 int K, int N) {
  int i = blockIdx.x * blockDim.x + threadIdx.x;
  if (i < K * N) {
    int n = i / K, k = i - n * K;
    out[(size_t)n * K + k] = (__bf16)in[(size_t)k * N + n];
  }
}

// ---------------------------------------------------------------- WMMA GEMM
// C[M,N] = A[M,K](bf16) * Bt[N,K](bf16)^T, fp32 accumulate.
// Double-buffered LDS; async global->LDS copies run one K-tile ahead of the WMMAs.
// EPI==0: plain fp32 store to out0.
// EPI==1: xz epilogue: cols [0,1024) -> x_main (out0), cols [1024,2048) -> silu -> z_silu (out1).
#define TILE_ELEMS (128 * 40)                   // padded 128x32 tile (80B row stride)
#define TILE_BYTES (TILE_ELEMS * 2)

template <int EPI>
__global__ __launch_bounds__(256)
void k_wmma_gemm(const __bf16* __restrict__ A, const __bf16* __restrict__ Bt,
                 float* __restrict__ out0, float* __restrict__ out1,
                 int M, int N, int K) {
  __shared__ __align__(16) __bf16 As[2][128][40];
  __shared__ __align__(16) __bf16 Bs[2][128][40];

  const int mBase = blockIdx.y * 128;
  const int nBase = blockIdx.x * 128;
  const int tid   = threadIdx.x;
  const int wave  = tid >> 5;
  const int lane  = tid & 31;
  const int half  = lane >> 4;      // which 16-lane half of the wave
  const int l16   = lane & 15;
  const int wm    = (wave >> 2) * 64;   // 2x4 wave grid: 64 rows x 32 cols per wave
  const int wn    = (wave & 3) * 32;

  // each thread owns two 16B chunks of each tile (512 chunks / 256 threads)
  const int c0 = tid, c1 = tid + 256;
  const int r0 = c0 >> 2, q0 = (c0 & 3) * 8;
  const int r1 = c1 >> 2, q1 = (c1 & 3) * 8;
  const unsigned ldsA0 = (unsigned)(uintptr_t)&As[0][r0][q0];
  const unsigned ldsA1 = (unsigned)(uintptr_t)&As[0][r1][q1];
  const unsigned ldsB0 = (unsigned)(uintptr_t)&Bs[0][r0][q0];
  const unsigned ldsB1 = (unsigned)(uintptr_t)&Bs[0][r1][q1];
  const __bf16* gA0 = A  + (size_t)(mBase + r0) * K + q0;
  const __bf16* gA1 = A  + (size_t)(mBase + r1) * K + q1;
  const __bf16* gB0 = Bt + (size_t)(nBase + r0) * K + q0;
  const __bf16* gB1 = Bt + (size_t)(nBase + r1) * K + q1;

  // issue this thread's 4 async 16B copies for one K-tile into LDS buffer `buf`
  auto stage = [&](int buf, int kb) {
    const unsigned o = (unsigned)buf * TILE_BYTES;
    asm volatile("global_load_async_to_lds_b128 %0, %1, off" :: "v"(ldsA0 + o), "v"(gA0 + kb) : "memory");
    asm volatile("global_load_async_to_lds_b128 %0, %1, off" :: "v"(ldsA1 + o), "v"(gA1 + kb) : "memory");
    asm volatile("global_load_async_to_lds_b128 %0, %1, off" :: "v"(ldsB0 + o), "v"(gB0 + kb) : "memory");
    asm volatile("global_load_async_to_lds_b128 %0, %1, off" :: "v"(ldsB1 + o), "v"(gB1 + kb) : "memory");
  };

  v8f acc[4][2];
#pragma unroll
  for (int i = 0; i < 4; ++i)
#pragma unroll
    for (int j = 0; j < 2; ++j)
#pragma unroll
      for (int r = 0; r < 8; ++r) acc[i][j][r] = 0.f;

  const int nk = K >> 5;            // K/32 tiles
  stage(0, 0);                      // prologue: fill buffer 0

  for (int it = 0; it < nk; ++it) {
    const int cur = it & 1;
    if (it + 1 < nk) {
      stage(cur ^ 1, (it + 1) * 32);          // copy next tile while we compute this one
      if (it + 2 < nk) {                      // warm L2 two tiles ahead (global_prefetch_b8)
        __builtin_prefetch(gA0 + (it + 2) * 32, 0, 3);
        __builtin_prefetch(gB0 + (it + 2) * 32, 0, 3);
      }
      // async loads retire in order: <=4 outstanding => the 4 older (current-tile) copies are done
      asm volatile("s_wait_asynccnt 4" ::: "memory");
    } else {
      asm volatile("s_wait_asynccnt 0" ::: "memory");
    }
    __syncthreads();   // current tile visible to all waves

    // B fragments: lane holds col (l16), K = half*16 + e (contiguous)
    v16bf bfrag[2];
#pragma unroll
    for (int j = 0; j < 2; ++j) {
      int nrow = wn + j * 16 + l16;
      v8bf b0 = *(const v8bf*)&Bs[cur][nrow][half * 16];
      v8bf b1 = *(const v8bf*)&Bs[cur][nrow][half * 16 + 8];
#pragma unroll
      for (int e = 0; e < 8; ++e) { bfrag[j][e] = b0[e]; bfrag[j][8 + e] = b1[e]; }
    }

#pragma unroll
    for (int i = 0; i < 4; ++i) {
      // A fragment: lane holds row (l16),
      // K = half*16 + e (contiguous, matching B)
      int mrow = wm + i * 16 + l16;
      v8bf a0 = *(const v8bf*)&As[cur][mrow][half * 16];
      v8bf a1 = *(const v8bf*)&As[cur][mrow][half * 16 + 8];
      v16bf afrag;
#pragma unroll
      for (int e = 0; e < 8; ++e) { afrag[e] = a0[e]; afrag[8 + e] = a1[e]; }
#pragma unroll
      for (int j = 0; j < 2; ++j) {
        acc[i][j] = __builtin_amdgcn_wmma_f32_16x16x32_bf16(
            false, afrag, false, bfrag[j], (short)0, acc[i][j], false, false);
      }
    }
    __syncthreads();   // all waves done reading `cur` before it is overwritten next+1 iter
  }

  // C/D layout: VGPR r -> M = r (lanes 0-15) / r+8 (lanes 16-31); N = lane%16
#pragma unroll
  for (int i = 0; i < 4; ++i)
#pragma unroll
    for (int j = 0; j < 2; ++j)
#pragma unroll
      for (int r = 0; r < 8; ++r) {
        int row = mBase + wm + i * 16 + r + half * 8;
        int col = nBase + wn + j * 16 + l16;
        float v = acc[i][j][r];
        if (EPI == 0) {
          out0[(size_t)row * N + col] = v;
        } else {
          if (col < D_MODELX) {
            out0[(size_t)row * D_MODELX + col] = v;               // x_main
          } else {
            float s = v * (1.f / (1.f + expf(-v)));               // silu(z)
            out1[(size_t)row * D_MODELX + (col - D_MODELX)] = s;  // z_silu
          }
        }
      }
}

// ---------------------------------------------------------------- conv + silu
__global__ void k_conv_silu(const float* __restrict__ xm, const float* __restrict__ cw,
                            const float* __restrict__ cb, float* __restrict__ xc) {
  int i = blockIdx.x * blockDim.x + threadIdx.x;
  if (i >= M_ROWS * D_MODELX) return;
  int d   = i & (D_MODELX - 1);
  int row = i >> 10;               // b*SEQ + t
  int t   = row & (SEQ_LENX - 1);
  float acc = cb[d];
#pragma unroll
  for (int j = 0; j < D_CONVX; ++j) {
    int tt = t - (D_CONVX - 1) + j;
    if (tt >= 0)
      acc += cw[d * D_CONVX + j] * xm[(size_t)(row - (D_CONVX - 1) + j) * D_MODELX + d];
  }
  xc[i] = acc * (1.f / (1.f + expf(-acc)));
}

// ---------------------------------------------------------------- x-projection
__global__ void k_xproj(const float* __restrict__ xc, const float* __restrict__ Wp,
                        float* __restrict__ Bsel, float* __restrict__ Csel,
                        float* __restrict__ delta) {
  int row = blockIdx.x;
  int j   = threadIdx.x;
  if (j >= 2 * D_STATEX + 1) return;
  const float* xr = xc + (size_t)row * D_MODELX;
  float acc = 0.f;
  for (int k = 0; k < D_MODELX; ++k)
    acc += xr[k] * Wp[k * (2 * D_STATEX + 1) + j];
  if (j < D_STATEX)            Bsel[row * D_STATEX + j] = acc;
  else if (j < 2 * D_STATEX)   Csel[row * D_STATEX + (j - D_STATEX)] = acc;
  else                         delta[row] = (acc > 20.f) ? acc : log1pf(expf(acc));  // softplus
}

// ---------------------------------------------------------------- selective scan
// one thread per (batch, channel); 16-wide state in registers; (delta,B,C) chunks in LDS
#define T_CHUNK 128
__global__ __launch_bounds__(256)
void k_scan(const float* __restrict__ xc, const float* __restrict__ zs,
            const float* __restrict__ Bsel, const float* __restrict__ Csel,
            const float* __restrict__ delta, const float* __restrict__ logA,
            const float* __restrict__ Dp, __bf16* __restrict__ ybf) {
  __shared__ float sD[T_CHUNK];
  __shared__ float sB[T_CHUNK][D_STATEX];
  __shared__ float sC[T_CHUNK][D_STATEX];

  int b = blockIdx.x >> 2;                              // 4 blocks per batch (1024/256)
  int d = ((blockIdx.x & 3) << 8) + threadIdx.x;

  float A[D_STATEX], h[D_STATEX];
#pragma unroll
  for (int n = 0; n < D_STATEX; ++n) {
    A[n] = -expf(logA[d * D_STATEX + n]);
    h[n] = 0.f;
  }
  float dpar = Dp[d];

  for (int t0 = 0; t0 < SEQ_LENX; t0 += T_CHUNK) {
    __syncthreads();
    for (int i = threadIdx.x; i < T_CHUNK; i += 256)
      sD[i] = delta[(size_t)b * SEQ_LENX + t0 + i];
    for (int i = threadIdx.x; i < T_CHUNK * D_STATEX; i += 256) {
      int tt = i >> 4, nn = i & 15;
      size_t base = ((size_t)b * SEQ_LENX + t0 + tt) * D_STATEX + nn;
      sB[tt][nn] = Bsel[base];
      sC[tt][nn] = Csel[base];
    }
    __syncthreads();

    for (int tc = 0; tc < T_CHUNK; ++tc) {
      size_t gi = ((size_t)b * SEQ_LENX + t0 + tc) * D_MODELX + d;
      float u   = xc[gi];
      float dlt = sD[tc];
      float du  = dlt * u;
      float y   = 0.f;
#pragma unroll
      for (int n = 0; n < D_STATEX; ++n) {
        h[n] = expf(dlt * A[n]) * h[n] + du * sB[tc][n];
        y += h[n] * sC[tc][n];
      }
      y += u * dpar;
      ybf[gi] = (__bf16)(y * zs[gi]);    // fuse silu(z) gating + bf16 convert for GEMM-2
    }
  }
}

// ---------------------------------------------------------------- launch

extern "C" void kernel_launch(void* const* d_in, const int* in_sizes, int n_in,
                              void* d_out, int out_size, void* d_ws, size_t ws_size,
                              hipStream_t stream) {
  const float* x    = (const float*)d_in[0];
  const float* Win  = (const float*)d_in[1];
  const float* cw   = (const float*)d_in[2];
  const float* cb   = (const float*)d_in[3];
  const float* Wp   = (const float*)d_in[4];
  const float* lA   = (const float*)d_in[5];
  const float* Dp   = (const float*)d_in[6];
  const float* Wout = (const float*)d_in[7];
  float* out = (float*)d_out;
  (void)in_sizes; (void)n_in; (void)out_size; (void)ws_size;

  char* ws = (char*)d_ws;
  size_t off = 0;
  auto wsAlloc = [&](size_t bytes) -> char* {
    char* p = ws + off;
    off += (bytes + 255) & ~(size_t)255;
    return p;
  };
  __bf16* x_bf   = (__bf16*)wsAlloc((size_t)M_ROWS * D_MODELX * 2);
  __bf16* Wt_in  = (__bf16*)wsAlloc((size_t)2 * D_MODELX * D_MODELX * 2);
  __bf16* Wt_out = (__bf16*)wsAlloc((size_t)D_MODELX * D_MODELX * 2);
  float*  x_main = (float*) wsAlloc((size_t)M_ROWS * D_MODELX * 4);
  float*  z_silu = (float*) wsAlloc((size_t)M_ROWS * D_MODELX * 4);
  float*  x_conv = (float*) wsAlloc((size_t)M_ROWS * D_MODELX * 4);
  __bf16* y_bf   = (__bf16*)wsAlloc((size_t)M_ROWS * D_MODELX * 2);
  float*  Bsel   = (float*) wsAlloc((size_t)M_ROWS * D_STATEX * 4);
  float*  Csel   = (float*) wsAlloc((size_t)M_ROWS * D_STATEX * 4);
  float*  delta  = (float*) wsAlloc((size_t)M_ROWS * 4);

  // prep: bf16 copies / transposed bf16 weights
  k_f32_to_bf16<<<(M_ROWS * D_MODELX + 255) / 256, 256, 0, stream>>>(x, x_bf, M_ROWS * D_MODELX);
  k_transpose_bf16<<<(2 * D_MODELX * D_MODELX + 255) / 256, 256, 0, stream>>>(Win, Wt_in, D_MODELX, 2 * D_MODELX);
  k_transpose_bf16<<<(D_MODELX * D_MODELX + 255) / 256, 256, 0, stream>>>(Wout, Wt_out, D_MODELX, D_MODELX);

  // GEMM 1: xz = x @ W_in  (fused split + silu(z))
  dim3 g1(2 * D_MODELX / 128, M_ROWS / 128);
  k_wmma_gemm<1><<<g1, 256, 0, stream>>>(x_bf, Wt_in, x_main, z_silu, M_ROWS, 2 * D_MODELX, D_MODELX);

  // depthwise causal conv + silu
  k_conv_silu<<<(M_ROWS * D_MODELX + 255) / 256, 256, 0, stream>>>(x_main, cw, cb, x_conv);

  // B/C/delta projection
  k_xproj<<<M_ROWS, 64, 0, stream>>>(x_conv, Wp, Bsel, Csel, delta);

  // selective scan (fused +u*D, *silu(z), bf16 convert)
  k_scan<<<BATCHX * (D_MODELX / 256), 256, 0, stream>>>(x_conv, z_silu, Bsel, Csel, delta, lA, Dp, y_bf);

  // GEMM 2: out = y @ W_out
  dim3 g2(D_MODELX / 128, M_ROWS / 128);
  k_wmma_gemm<0><<<g2, 256, 0, stream>>>(y_bf, Wt_out, out, nullptr, M_ROWS, D_MODELX, D_MODELX);
}
